// OperandExtractor_43928925504167
// MI455X (gfx1250) — matrix operand order repaired and checked by reading, not compiled, not run on hardware
//
#include <hip/hip_runtime.h>
#include <cstddef>
#include <cstdint>

// Problem constants (match reference).
static constexpr int Bn = 4, Sn = 2048, Hn = 2048, Rn = 32, Mh = 128, Od = 100;
#define NEGV (-1e9f)
#define SCALE_QK (0.17677669529663687f) /* 1/sqrt(32) */

typedef __attribute__((ext_vector_type(16))) __bf16 v16bf;
typedef __attribute__((ext_vector_type(8)))  float  v8f;
typedef __attribute__((ext_vector_type(16))) float  v16f;
typedef __attribute__((ext_vector_type(16))) unsigned short v16u;
typedef __attribute__((ext_vector_type(8)))  unsigned short v8u;

// Native hardware f32->bf16 convert (RNE).
__device__ __forceinline__ void st_bf(unsigned short* p, float f) {
  *(__bf16*)p = (__bf16)f;
}

__device__ __forceinline__ v16bf mkbf(v16u u) { return __builtin_bit_cast(v16bf, u); }

// ---- CDNA5 async global->LDS copy (GLOBAL_LOAD_ASYNC_TO_LDS_B128, ASYNCcnt) ----
typedef __attribute__((address_space(3))) unsigned short lds_u16_t;
__device__ __forceinline__ unsigned lds_addr(const unsigned short* p) {
  return (unsigned)(uintptr_t)(lds_u16_t*)(unsigned short*)p;
}
__device__ __forceinline__ void async_copy_b128(unsigned lds_dst, const void* gsrc) {
  asm volatile("global_load_async_to_lds_b128 %0, %1, off"
               :: "v"(lds_dst), "v"((unsigned long long)(uintptr_t)gsrc)
               : "memory");
}
__device__ __forceinline__ void wait_asynccnt0() {
  asm volatile("s_wait_asynccnt 0x0" ::: "memory");
}

// ---- WMMA operand loaders (16-bit, contiguous k order k = 16h + e for A and B, wave32) ----
// A (M=16 x K=32): lane L holds row m=L&15; element e -> k = (L>>4)*16 + e
__device__ __forceinline__ v16bf load_a_bf16(const unsigned short* p, int ld, int lane) {
  int m = lane & 15, half = lane >> 4;
  const unsigned short* r = p + (size_t)m * ld + half * 16;
  v8u c0 = *(const v8u*)r;
  v8u c1 = *(const v8u*)(r + 8);
  v16u u = __builtin_shufflevector(c0, c1, 0, 1, 2, 3, 4, 5, 6, 7,
                                   8, 9, 10, 11, 12, 13, 14, 15);
  return mkbf(u);
}

__device__ __forceinline__ v16bf load_a_f32(const float* p, int ld, int lane) {
  int m = lane & 15, half = lane >> 4;
  const float* r = p + (size_t)m * ld + half * 16;
  v8f c0 = *(const v8f*)r;
  v8f c1 = *(const v8f*)(r + 8);
  v16bf u;
#pragma unroll
  for (int i = 0; i < 8; ++i) { u[i] = (__bf16)c0[i]; u[8 + i] = (__bf16)c1[i]; }
  return u;
}

// B (K=32 x N=16) with B[k][n] = src[n][k]: lane holds col n=L&15; e -> k = (L>>4)*16 + e
__device__ __forceinline__ v16bf load_b_bf16(const unsigned short* p, int ld, int lane) {
  int n = lane & 15, half = lane >> 4;
  return mkbf(*(const v16u*)(p + (size_t)n * ld + half * 16));
}

// Same but via two 16B loads (for LDS rows with 16B-but-not-32B alignment)
__device__ __forceinline__ v16bf load_b_bf16_2x8(const unsigned short* p, int ld, int lane) {
  int n = lane & 15, half = lane >> 4;
  const unsigned short* r = p + (size_t)n * ld + half * 16;
  v8u c0 = *(const v8u*)r;
  v8u c1 = *(const v8u*)(r + 8);
  v16u u = __builtin_shufflevector(c0, c1, 0, 1, 2, 3, 4, 5, 6, 7,
                                   8, 9, 10, 11, 12, 13, 14, 15);
  return mkbf(u);
}

__device__ __forceinline__ v16bf load_b_f32(const float* p, int ld, int lane) {
  int n = lane & 15, half = lane >> 4;
  v16f v = *(const v16f*)(p + (size_t)n * ld + half * 16);
  v16bf u;
#pragma unroll
  for (int i = 0; i < 16; ++i) u[i] = (__bf16)v[i];
  return u;
}

__device__ __forceinline__ v16bf load_b_f32_rowptr(const float* rowk, int half) {
  v16f v = *(const v16f*)(rowk + half * 16);
  v16bf u;
#pragma unroll
  for (int i = 0; i < 16; ++i) u[i] = (__bf16)v[i];
  return u;
}

__device__ __forceinline__ v8f wmma_bf16(v16bf a, v16bf b, v8f c) {
  return __builtin_amdgcn_wmma_f32_16x16x32_bf16(false, a, false, b, (short)0, c, false, false);
}

__device__ __forceinline__ float redmax16(float v) {
  v = fmaxf(v, __shfl_xor(v, 8, 32));
  v = fmaxf(v, __shfl_xor(v, 4, 32));
  v = fmaxf(v, __shfl_xor(v, 2, 32));
  v = fmaxf(v, __shfl_xor(v, 1, 32));
  return v;
}
__device__ __forceinline__ float redsum16(float v) {
  v += __shfl_xor(v, 8, 32);
  v += __shfl_xor(v, 4, 32);
  v += __shfl_xor(v, 2, 32);
  v += __shfl_xor(v, 1, 32);
  return v;
}

// ---- Kernel 1: h[B][S][H] f32 -> hT[B][H][S] bf16 (for PV B-operand) ----
__global__ void k_transpose(const float* __restrict__ h, unsigned short* __restrict__ hT) {
  __shared__ __align__(16) float t[32][33];
  int b = blockIdx.z;
  int s0 = blockIdx.y * 32, h0 = blockIdx.x * 32;
  int tx = threadIdx.x, ty = threadIdx.y;
#pragma unroll
  for (int r = ty; r < 32; r += 8)
    t[r][tx] = h[((size_t)(b * Sn + s0 + r)) * Hn + h0 + tx];
  __syncthreads();
#pragma unroll
  for (int r = ty; r < 32; r += 8)
    st_bf(&hT[((size_t)(b * Hn + h0 + r)) * Sn + s0 + tx], t[tx][r]);
}

// ---- Kernel 2: Q or K projection: out[b][s][r] = sum_h h[b][s][h]*W[r][h] (bf16 out) ----
__global__ __launch_bounds__(32) void k_proj(const float* __restrict__ h,
                                             const float* __restrict__ W,
                                             unsigned short* __restrict__ out) {
  int lane = threadIdx.x;
  int b = blockIdx.y, s0 = blockIdx.x * 16;
  int n = lane & 15, half = lane >> 4;
  v8f acc0 = {}, acc1 = {};
  const float* hrow = h + ((size_t)(b * Sn + s0)) * Hn;
  for (int k0 = 0; k0 < Hn; k0 += 32) {
    v16bf a  = load_a_f32(hrow + k0, Hn, lane);
    v16bf w0 = load_b_f32(W + k0, Hn, lane);                    // rows 0..15
    v16bf w1 = load_b_f32(W + (size_t)16 * Hn + k0, Hn, lane);  // rows 16..31
    acc0 = wmma_bf16(a, w0, acc0);
    acc1 = wmma_bf16(a, w1, acc1);
  }
  unsigned short* o = out + ((size_t)(b * Sn + s0 + half * 8)) * Rn;
#pragma unroll
  for (int i = 0; i < 8; ++i) {
    st_bf(&o[(size_t)i * Rn + n], acc0[i]);
    st_bf(&o[(size_t)i * Rn + 16 + n], acc1[i]);
  }
}

// ---- Kernel 3: online softmax statistics (row max, row sum) ----
__global__ __launch_bounds__(32) void k_stats(const unsigned short* __restrict__ Q,
                                              const unsigned short* __restrict__ K,
                                              const int* __restrict__ amask,
                                              float* __restrict__ omax,
                                              float* __restrict__ osum) {
  int lane = threadIdx.x;
  int b = blockIdx.y, q0 = blockIdx.x * 16;
  int n = lane & 15, half = lane >> 4;
  v16bf a = load_a_bf16(Q + ((size_t)(b * Sn + q0)) * Rn, Rn, lane);
  float mrun[8], lrun[8];
#pragma unroll
  for (int i = 0; i < 8; ++i) { mrun[i] = -3.0e38f; lrun[i] = 0.f; }
  for (int t0 = 0; t0 < q0 + 16; t0 += 16) {
    v16bf kb = load_b_bf16(K + ((size_t)(b * Sn + t0)) * Rn, Rn, lane);
    v8f z = {};
    v8f sv = wmma_bf16(a, kb, z);
    int t = t0 + n;
    bool mval = (amask[b * Sn + t] != 0);
#pragma unroll
    for (int i = 0; i < 8; ++i) {
      int q = q0 + half * 8 + i;
      float s = (t <= q && mval) ? sv[i] * SCALE_QK : NEGV;
      float tm   = redmax16(s);
      float mnew = fmaxf(mrun[i], tm);
      float p    = expf(s - mnew);
      float ts   = redsum16(p);
      lrun[i] = lrun[i] * expf(mrun[i] - mnew) + ts;
      mrun[i] = mnew;
    }
  }
  if (n == 0) {
#pragma unroll
    for (int i = 0; i < 8; ++i) {
      omax[b * Sn + q0 + half * 8 + i] = mrun[i];
      osum[b * Sn + q0 + half * 8 + i] = lrun[i];
    }
  }
}

// ---- Kernel 4: att = softmax(QK^T) @ h ; bf16 out.
// 8 waves / block; 128 queries x 128 H-cols per block; V tile staged in LDS via
// async global->LDS copies, double-buffered; all waves share the V tile.
__global__ __launch_bounds__(256) void k_pv(const unsigned short* __restrict__ Q,
                                            const unsigned short* __restrict__ K,
                                            const unsigned short* __restrict__ hT,
                                            const int* __restrict__ amask,
                                            const float* __restrict__ imax,
                                            const float* __restrict__ isum,
                                            unsigned short* __restrict__ att) {
  // V tile: [col 0..127][key 0..31] bf16, row stride 40 u16 (80B, 16B-aligned rows)
  __shared__ __align__(16) unsigned short Vt[2][128][40];
  __shared__ __align__(16) unsigned short Pp[8][16][40];  // per-wave prob tiles
  int tid = threadIdx.x;
  int lane = tid & 31, w = tid >> 5;
  int b = blockIdx.z, c0 = blockIdx.y * 128, qb = blockIdx.x * 128;
  int q0 = qb + w * 16;
  int n = lane & 15, half = lane >> 4;

  v16bf a = load_a_bf16(Q + ((size_t)(b * Sn + q0)) * Rn, Rn, lane);
  float rmx[8], rsi[8];
#pragma unroll
  for (int i = 0; i < 8; ++i) {
    rmx[i] = imax[b * Sn + q0 + half * 8 + i];
    rsi[i] = 1.0f / isum[b * Sn + q0 + half * 8 + i];
  }
  v8f O[8];
#pragma unroll
  for (int j = 0; j < 8; ++j) O[j] = {};

  const unsigned short* hTb = hT + ((size_t)(b * Hn + c0)) * Sn;
  const int tmax = qb + 128;

  // Prefetch first V tile: 128 cols x 32 keys (64B/row), 16B per thread-chunk.
  {
#pragma unroll
    for (int it = 0; it < 2; ++it) {
      int idx = it * 256 + tid;
      int c = idx >> 2, ch = idx & 3;
      async_copy_b128(lds_addr(&Vt[0][c][ch * 8]),
                      hTb + (size_t)c * Sn + 0 + ch * 8);
    }
  }

  int buf = 0;
  for (int t0 = 0; t0 < tmax; t0 += 32) {
    wait_asynccnt0();       // this wave's async copies into Vt[buf] complete
    __syncthreads();        // publish Vt[buf]; previous reads of Vt[buf^1] done
    if (t0 + 32 < tmax) {
#pragma unroll
      for (int it = 0; it < 2; ++it) {
        int idx = it * 256 + tid;
        int c = idx >> 2, ch = idx & 3;
        async_copy_b128(lds_addr(&Vt[buf ^ 1][c][ch * 8]),
                        hTb + (size_t)c * Sn + (t0 + 32) + ch * 8);
      }
    }
    if (t0 < q0 + 16) {     // uniform per wave (causal limit)
#pragma unroll
      for (int sub = 0; sub < 2; ++sub) {
        int tt = t0 + sub * 16;
        if (tt < q0 + 16) {
          v16bf kb = load_b_bf16(K + ((size_t)(b * Sn + tt)) * Rn, Rn, lane);
          v8f z = {};
          v8f sv = wmma_bf16(a, kb, z);
          int t = tt + n;
          bool mval = (amask[b * Sn + t] != 0);
#pragma unroll
          for (int i = 0; i < 8; ++i) {
            int q = q0 + half * 8 + i;
            float p = 0.f;
            if (t <= q && mval) p = expf(sv[i] * SCALE_QK - rmx[i]) * rsi[i];
            st_bf(&Pp[w][half * 8 + i][sub * 16 + n], p);
          }
        } else {
#pragma unroll
          for (int i = 0; i < 8; ++i) Pp[w][half * 8 + i][sub * 16 + n] = 0;
        }
      }
      v16bf pa = load_a_bf16(&Pp[w][0][0], 40, lane);   // own-wave LDS, in-order
#pragma unroll
      for (int j = 0; j < 8; ++j) {
        v16bf bb = load_b_bf16_2x8(&Vt[buf][j * 16][0], 40, lane);
        O[j] = wmma_bf16(pa, bb, O[j]);
      }
    }
    buf ^= 1;
  }

  unsigned short* ob = att + ((size_t)(b * Sn + q0 + half * 8)) * Hn + c0;
#pragma unroll
  for (int j = 0; j < 8; ++j)
#pragma unroll
    for (int i = 0; i < 8; ++i)
      st_bf(&ob[(size_t)i * Hn + j * 16 + n], O[j][i]);
}

// ---- Kernel 5: MLP head -> logits[b][s][100] fp32 ----
__global__ __launch_bounds__(32) void k_mlp(const unsigned short* __restrict__ att,
                                            const float* __restrict__ W1,
                                            const float* __restrict__ b1,
                                            const float* __restrict__ W2,
                                            const float* __restrict__ b2,
                                            float* __restrict__ logits) {
  __shared__ __align__(16) unsigned short Y[16][136];
  int lane = threadIdx.x;
  int b = blockIdx.y, s0 = blockIdx.x * 16;
  int n = lane & 15, half = lane >> 4;

  v8f acc[8];
#pragma unroll
  for (int j = 0; j < 8; ++j) acc[j] = {};
  const unsigned short* arow = att + ((size_t)(b * Sn + s0)) * Hn;
  for (int k0 = 0; k0 < Hn; k0 += 32) {
    v16bf a = load_a_bf16(arow + k0, Hn, lane);
#pragma unroll
    for (int j = 0; j < 8; ++j) {
      v16bf wv = load_b_f32(W1 + (size_t)(j * 16) * Hn + k0, Hn, lane);
      acc[j] = wmma_bf16(a, wv, acc[j]);
    }
  }
#pragma unroll
  for (int j = 0; j < 8; ++j) {
    float bias = b1[j * 16 + n];
#pragma unroll
    for (int i = 0; i < 8; ++i) {
      float x = acc[j][i] + bias;
      float g = 0.5f * x * (1.0f + erff(x * 0.70710678118654752f));
      st_bf(&Y[half * 8 + i][j * 16 + n], g);
    }
  }
  __syncthreads();

  v8f a2[7];
#pragma unroll
  for (int j = 0; j < 7; ++j) a2[j] = {};
#pragma unroll
  for (int k0 = 0; k0 < Mh; k0 += 32) {
    v16bf a = load_a_bf16(&Y[0][k0], 136, lane);
#pragma unroll
    for (int j = 0; j < 7; ++j) {
      int row = j * 16 + n;                 // output channel (pad 100 -> 112)
      v16bf wv;
      if (row < Od) {
        wv = load_b_f32_rowptr(W2 + (size_t)row * Mh + k0, half);
      } else {
        v16u z = {};
        wv = mkbf(z);
      }
      a2[j] = wmma_bf16(a, wv, a2[j]);
    }
  }
  float* lg = logits + ((size_t)(b * Sn + s0 + half * 8)) * Od;
#pragma unroll
  for (int j = 0; j < 7; ++j) {
    int col = j * 16 + n;
    if (col < Od) {
      float bias = b2[col];
#pragma unroll
      for (int i = 0; i < 8; ++i)
        lg[(size_t)i * Od + col] = a2[j][i] + bias;
    }
  }
}

// ---- Kernel 6: STE argmax forward == hard argmax ----
__global__ void k_digits(const float* __restrict__ logits, float* __restrict__ d) {
  int idx = blockIdx.x * blockDim.x + threadIdx.x;
  if (idx >= Bn * Sn * 10) return;
  const float* p = logits + (size_t)(idx / 10) * Od + (idx % 10) * 10;
  int best = 0;
  float bv = p[0];
#pragma unroll
  for (int c = 1; c < 10; ++c) {
    float v = p[c];
    if (v > bv) { bv = v; best = c; }
  }
  d[idx] = (float)best;
}

extern "C" void kernel_launch(void* const* d_in, const int* in_sizes, int n_in,
                              void* d_out, int out_size, void* d_ws, size_t ws_size,
                              hipStream_t stream) {
  (void)in_sizes; (void)n_in; (void)out_size; (void)ws_size;
  const float* h     = (const float*)d_in[0];
  const int*   amask = (const int*)d_in[1];
  const float* Wq[2] = {(const float*)d_in[2], (const float*)d_in[4]};
  const float* Wk[2] = {(const float*)d_in[3], (const float*)d_in[5]};
  const float* W1[2] = {(const float*)d_in[6], (const float*)d_in[10]};
  const float* B1[2] = {(const float*)d_in[7], (const float*)d_in[11]};
  const float* W2[2] = {(const float*)d_in[8], (const float*)d_in[12]};
  const float* B2[2] = {(const float*)d_in[9], (const float*)d_in[13]};

  // Workspace layout (bytes): hT 32MB | att 32MB | Q 512KB | K 512KB | rmax 32KB | rsum 32KB
  char* ws = (char*)d_ws;
  unsigned short* hT  = (unsigned short*)(ws);
  unsigned short* att = (unsigned short*)(ws + 33554432);
  unsigned short* Qw  = (unsigned short*)(ws + 67108864);
  unsigned short* Kw  = (unsigned short*)(ws + 67633152);
  float* rmx = (float*)(ws + 68157440);
  float* rsm = (float*)(ws + 68190208);

  // Output: d_a[81920] | d_b[81920] | logits_a[819200] | logits_b[819200]
  float* out = (float*)d_out;
  float* Dd[2] = {out, out + 81920};
  float* LG[2] = {out + 163840, out + 983040};

  k_transpose<<<dim3(Hn / 32, Sn / 32, Bn), dim3(32, 8, 1), 0, stream>>>(h, hT);

  for (int hd = 0; hd < 2; ++hd) {
    k_proj  <<<dim3(Sn / 16, Bn), 32, 0, stream>>>(h, Wq[hd], Qw);
    k_proj  <<<dim3(Sn / 16, Bn), 32, 0, stream>>>(h, Wk[hd], Kw);
    k_stats <<<dim3(Sn / 16, Bn), 32, 0, stream>>>(Qw, Kw, amask, rmx, rsm);
    k_pv    <<<dim3(Sn / 128, Hn / 128, Bn), 256, 0, stream>>>(Qw, Kw, hT, amask, rmx, rsm, att);
    k_mlp   <<<dim3(Sn / 16, Bn), 32, 0, stream>>>(att, W1[hd], B1[hd], W2[hd], B2[hd], LG[hd]);
    k_digits<<<dim3((Bn * Sn * 10 + 255) / 256), 256, 0, stream>>>(LG[hd], Dd[hd]);
  }
}
